// Decoder_58377195487266
// MI455X (gfx1250) — matrix operand order repaired and checked by reading, not compiled, not run on hardware
//
#include <hip/hip_runtime.h>
#include <math.h>

typedef __bf16 bf16;
typedef __attribute__((ext_vector_type(16))) __bf16 v16bf;
typedef __attribute__((ext_vector_type(8)))  __bf16 v8bf;
typedef __attribute__((ext_vector_type(8)))  float  v8f;

#define B_   16
#define T_   2048
#define E_   256
#define DK_  32
#define H_   8
#define BT_  (B_ * T_)
#define FF_  (4 * E_)          // 1024
#define SCALE_ 0.17677669529663687f   // 1/sqrt(32)
#define LN_EPS_ 1e-5f

static __device__ __forceinline__ v8f wmma_bf16(v16bf a, v16bf b, v8f c) {
  return __builtin_amdgcn_wmma_f32_16x16x32_bf16(false, a, false, b, (short)0, c,
                                                 false, false);
}
static __device__ __forceinline__ v8f zero8() {
  v8f c;
#pragma unroll
  for (int i = 0; i < 8; ++i) c[i] = 0.0f;
  return c;
}

// A operand (16x32, bf16) from row-major bf16 [rows][ld], row tile r0, K tile k0.
// Lane l: row = l&15; chunks K = [k0+8*hi, +8) and [k0+16+8*hi, +8).
static __device__ __forceinline__ v16bf load_a_bf16(const bf16* src, int ld,
                                                    int r0, int k0) {
  const int l  = threadIdx.x & 31;
  const int hi = l >> 4;
  const bf16* p = src + (size_t)(r0 + (l & 15)) * ld + k0 + 8 * hi;
  union { v16bf v; v8bf h[2]; } u;
  u.h[0] = *(const v8bf*)(p);
  u.h[1] = *(const v8bf*)(p + 16);
  return u.v;
}

// Same but converting from f32 source on the fly.
static __device__ __forceinline__ v16bf load_a_f32(const float* src, int ld,
                                                   int r0, int k0) {
  const int l  = threadIdx.x & 31;
  const int hi = l >> 4;
  const float* p = src + (size_t)(r0 + (l & 15)) * ld + k0 + 8 * hi;
  union { v16bf v; v8bf h[2]; } u;
#pragma unroll
  for (int i = 0; i < 8; ++i) u.h[0][i] = (bf16)p[i];
#pragma unroll
  for (int i = 0; i < 8; ++i) u.h[1][i] = (bf16)p[16 + i];
  return u.v;
}

// B operand (32x16, bf16) from TRANSPOSED matrix bt = B^T, row-major [N][ldk].
// Lane l: column n = n0 + (l&15); chunks K = [k0+8*hi, +8) and [k0+16+8*hi, +8).
static __device__ __forceinline__ v16bf load_bt(const bf16* bt, int ldk,
                                                int n0, int k0) {
  const int l = threadIdx.x & 31;
  const bf16* p = bt + (size_t)(n0 + (l & 15)) * ldk + k0 + 8 * (l >> 4);
  union { v16bf v; v8bf h[2]; } u;
  u.h[0] = *(const v8bf*)(p);
  u.h[1] = *(const v8bf*)(p + 16);
  return u.v;
}

// ---------------- weight prep ----------------
// dst[n*K+k] = (bf16) src[k*N+n]
__global__ void k_transpose(const float* __restrict__ src, bf16* __restrict__ dst,
                            int K, int N) {
  int idx = blockIdx.x * blockDim.x + threadIdx.x;
  if (idx >= K * N) return;
  int n = idx / K, k = idx % K;
  dst[idx] = (bf16)src[(size_t)k * N + n];
}

// WpfT[j][d] = sum_h Wp[h*32+d][j]   (fold of the head-tiling, then transpose)
__global__ void k_fold_wp(const float* __restrict__ Wp, bf16* __restrict__ WpfT) {
  int idx = blockIdx.x * blockDim.x + threadIdx.x;
  if (idx >= E_ * DK_) return;
  int j = idx / DK_, d = idx % DK_;
  float s = 0.0f;
#pragma unroll
  for (int h = 0; h < H_; ++h) s += Wp[(size_t)(h * DK_ + d) * E_ + j];
  WpfT[idx] = (bf16)s;
}

// ---------------- QKV projection ----------------
// One wave per 16-row tile. q,k row-major bf16 [BT][32]; v stored transposed
// vT[b][d][t] (transposed store of the C layout is per-lane contiguous).
__global__ void k_qkv(const float* __restrict__ x,
                      const bf16* __restrict__ WqT, const bf16* __restrict__ WkT,
                      const bf16* __restrict__ WvT,
                      bf16* __restrict__ q, bf16* __restrict__ k,
                      bf16* __restrict__ vT) {
  const int wave = (blockIdx.x * blockDim.x + threadIdx.x) >> 5;
  const int t0 = wave * 16;
  const int l = threadIdx.x & 31, hi = l >> 4, n = l & 15;
  v8f cq0 = zero8(), cq1 = zero8(), ck0 = zero8(), ck1 = zero8(),
      cv0 = zero8(), cv1 = zero8();
  for (int k0 = 0; k0 < E_; k0 += 32) {
    v16bf a = load_a_f32(x, E_, t0, k0);
    cq0 = wmma_bf16(a, load_bt(WqT, E_, 0,  k0), cq0);
    cq1 = wmma_bf16(a, load_bt(WqT, E_, 16, k0), cq1);
    ck0 = wmma_bf16(a, load_bt(WkT, E_, 0,  k0), ck0);
    ck1 = wmma_bf16(a, load_bt(WkT, E_, 16, k0), ck1);
    cv0 = wmma_bf16(a, load_bt(WvT, E_, 0,  k0), cv0);
    cv1 = wmma_bf16(a, load_bt(WvT, E_, 16, k0), cv1);
  }
#pragma unroll
  for (int j = 0; j < 8; ++j) {
    int t = t0 + j + 8 * hi;
    q[(size_t)t * DK_ + n]      = (bf16)cq0[j];
    q[(size_t)t * DK_ + 16 + n] = (bf16)cq1[j];
    k[(size_t)t * DK_ + n]      = (bf16)ck0[j];
    k[(size_t)t * DK_ + 16 + n] = (bf16)ck1[j];
  }
  // transposed contiguous store of V
  const int b  = t0 / T_;
  const int tl = (t0 - b * T_) + 8 * hi;
  v8bf s0, s1;
#pragma unroll
  for (int j = 0; j < 8; ++j) { s0[j] = (bf16)cv0[j]; s1[j] = (bf16)cv1[j]; }
  *(v8bf*)(vT + ((size_t)b * DK_ + n)      * T_ + tl) = s0;
  *(v8bf*)(vT + ((size_t)b * DK_ + 16 + n) * T_ + tl) = s1;
}

// ---------------- causal flash attention ----------------
#define FA_WAVES 4
__global__ void k_attn(const bf16* __restrict__ q, const bf16* __restrict__ kk,
                       const bf16* __restrict__ vT, bf16* __restrict__ attn) {
  __shared__ bf16 Pls[FA_WAVES][16][32];
  const int wv = threadIdx.x >> 5;
  const int wave = blockIdx.x * FA_WAVES + wv;
  const int t0g = wave * 16;
  const int b = t0g / T_;
  const int t0 = t0g - b * T_;
  const int l = threadIdx.x & 31, hi = l >> 4, n = l & 15;
  const bf16* kb = kk + (size_t)b * T_ * DK_;
  const bf16* vb = vT + (size_t)b * DK_ * T_;

  const v16bf aQ = load_a_bf16(q, DK_, t0g, 0);
  float mrow[8], lrow[8];
  v8f o0 = zero8(), o1 = zero8();
#pragma unroll
  for (int j = 0; j < 8; ++j) { mrow[j] = -3.0e38f; lrow[j] = 0.0f; }

  const int smax = ((t0 + 15) / 32) * 32;
  for (int s0 = 0; s0 <= smax; s0 += 32) {
    v8f st0 = wmma_bf16(aQ, load_bt(kb, DK_, s0,      0), zero8());
    v8f st1 = wmma_bf16(aQ, load_bt(kb, DK_, s0 + 16, 0), zero8());
    const bool maskblk = (s0 + 31 > t0);
    float p0[8], p1[8];
#pragma unroll
  for (int j = 0; j < 8; ++j) {
      const int t = t0 + j + 8 * hi;
      float v0 = st0[j] * SCALE_;
      float v1 = st1[j] * SCALE_;
      if (maskblk) {
        if (s0 + n > t)      v0 = -3.0e38f;
        if (s0 + 16 + n > t) v1 = -3.0e38f;
      }
      float mc = fmaxf(v0, v1);
#pragma unroll
      for (int d = 1; d < 16; d <<= 1) mc = fmaxf(mc, __shfl_xor(mc, d, 32));
      const float mn   = fmaxf(mrow[j], mc);
      const float corr = expf(mrow[j] - mn);
      const float e0 = expf(v0 - mn);
      const float e1 = expf(v1 - mn);
      float ls = e0 + e1;
#pragma unroll
      for (int d = 1; d < 16; d <<= 1) ls += __shfl_xor(ls, d, 32);
      lrow[j] = lrow[j] * corr + ls;
      mrow[j] = mn;
      o0[j] *= corr;
      o1[j] *= corr;
      p0[j] = e0;
      p1[j] = e1;
    }
    // C-layout -> LDS -> A-layout for P
#pragma unroll
    for (int j = 0; j < 8; ++j) {
      Pls[wv][j + 8 * hi][n]      = (bf16)p0[j];
      Pls[wv][j + 8 * hi][16 + n] = (bf16)p1[j];
    }
    asm volatile("s_wait_dscnt 0" ::: "memory");  // wave-internal DS RAW
    union { v16bf v; v8bf h[2]; } uP;
    uP.h[0] = *(const v8bf*)&Pls[wv][n][8 * hi];
    uP.h[1] = *(const v8bf*)&Pls[wv][n][16 + 8 * hi];
    o0 = wmma_bf16(uP.v, load_bt(vb, T_, 0,  s0), o0);
    o1 = wmma_bf16(uP.v, load_bt(vb, T_, 16, s0), o1);
  }
#pragma unroll
  for (int j = 0; j < 8; ++j) {
    const int t = t0g + j + 8 * hi;
    const float inv = 1.0f / lrow[j];
    attn[(size_t)t * DK_ + n]      = (bf16)(o0[j] * inv);
    attn[(size_t)t * DK_ + 16 + n] = (bf16)(o1[j] * inv);
  }
}

// ---------------- proj (folded Wp) + bias + residual + LayerNorm1 ----------------
// Block = 8 waves, owns a full 16x256 row tile; wave w covers cols [32w,32w+32).
__global__ void k_mh_ln(const bf16* __restrict__ attn, const bf16* __restrict__ WpfT,
                        const float* __restrict__ bp, const float* __restrict__ x,
                        const float* __restrict__ g1, const float* __restrict__ be1,
                        float* __restrict__ y1) {
  __shared__ float yt[16][E_ + 4];
  const int t0 = blockIdx.x * 16;
  const int w = threadIdx.x >> 5, l = threadIdx.x & 31, hi = l >> 4, n = l & 15;
  const int nb = 32 * w;
  v16bf a = load_a_bf16(attn, DK_, t0, 0);
  v8f c0 = wmma_bf16(a, load_bt(WpfT, DK_, nb,      0), zero8());
  v8f c1 = wmma_bf16(a, load_bt(WpfT, DK_, nb + 16, 0), zero8());
  const int col0 = nb + n, col1 = nb + 16 + n;
  const float bp0 = bp[col0], bp1 = bp[col1];
#pragma unroll
  for (int j = 0; j < 8; ++j) {
    const int m = j + 8 * hi;
    const int t = t0 + m;
    yt[m][col0] = c0[j] + bp0 + x[(size_t)t * E_ + col0];
    yt[m][col1] = c1[j] + bp1 + x[(size_t)t * E_ + col1];
  }
  __syncthreads();
#pragma unroll
  for (int rr = 0; rr < 2; ++rr) {
    const int r = 2 * w + rr;
    const int t = t0 + r;
    const int cb = l * 8;
    float vals[8], s = 0.0f, ss = 0.0f;
#pragma unroll
    for (int i = 0; i < 8; ++i) {
      vals[i] = yt[r][cb + i];
      s += vals[i];
      ss += vals[i] * vals[i];
    }
#pragma unroll
    for (int d = 1; d < 32; d <<= 1) { s += __shfl_xor(s, d, 32); ss += __shfl_xor(ss, d, 32); }
    const float mean = s * (1.0f / E_);
    const float var  = ss * (1.0f / E_) - mean * mean;
    const float rstd = rsqrtf(var + LN_EPS_);
#pragma unroll
    for (int i = 0; i < 8; ++i) {
      const int c = cb + i;
      y1[(size_t)t * E_ + c] = (vals[i] - mean) * rstd * g1[c] + be1[c];
    }
  }
}

// ---------------- FFN layer 1: gelu(y1 @ W1 + b1) -> bf16 ----------------
__global__ void k_ff1(const float* __restrict__ y1, const bf16* __restrict__ W1T,
                      const float* __restrict__ b1, bf16* __restrict__ h1) {
  const int mt = blockIdx.x >> 2;
  const int nseg = (blockIdx.x & 3) * 256;
  const int w = threadIdx.x >> 5, l = threadIdx.x & 31, hi = l >> 4, n = l & 15;
  const int t0 = mt * 16;
  const int nb = nseg + 32 * w;
  v8f c0 = zero8(), c1 = zero8();
  for (int k0 = 0; k0 < E_; k0 += 32) {
    v16bf a = load_a_f32(y1, E_, t0, k0);
    c0 = wmma_bf16(a, load_bt(W1T, E_, nb,      k0), c0);
    c1 = wmma_bf16(a, load_bt(W1T, E_, nb + 16, k0), c1);
  }
  const int col0 = nb + n, col1 = nb + 16 + n;
  const float bb0 = b1[col0], bb1 = b1[col1];
#pragma unroll
  for (int j = 0; j < 8; ++j) {
    const int t = t0 + j + 8 * hi;
    float u0 = c0[j] + bb0;
    float u1 = c1[j] + bb1;
    u0 = 0.5f * u0 * (1.0f + erff(u0 * 0.70710678118654752f));
    u1 = 0.5f * u1 * (1.0f + erff(u1 * 0.70710678118654752f));
    h1[(size_t)t * FF_ + col0] = (bf16)u0;
    h1[(size_t)t * FF_ + col1] = (bf16)u1;
  }
}

// ---------------- FFN layer 2 + bias + residual + LayerNorm2 -> out ----------------
__global__ void k_ff2_ln(const bf16* __restrict__ h1, const bf16* __restrict__ W2T,
                         const float* __restrict__ b2, const float* __restrict__ y1,
                         const float* __restrict__ g2, const float* __restrict__ be2,
                         float* __restrict__ out) {
  __shared__ float yt[16][E_ + 4];
  const int t0 = blockIdx.x * 16;
  const int w = threadIdx.x >> 5, l = threadIdx.x & 31, hi = l >> 4, n = l & 15;
  const int nb = 32 * w;
  v8f c0 = zero8(), c1 = zero8();
  for (int k0 = 0; k0 < FF_; k0 += 32) {
    v16bf a = load_a_bf16(h1, FF_, t0, k0);
    c0 = wmma_bf16(a, load_bt(W2T, FF_, nb,      k0), c0);
    c1 = wmma_bf16(a, load_bt(W2T, FF_, nb + 16, k0), c1);
  }
  const int col0 = nb + n, col1 = nb + 16 + n;
  const float bb0 = b2[col0], bb1 = b2[col1];
#pragma unroll
  for (int j = 0; j < 8; ++j) {
    const int m = j + 8 * hi;
    const int t = t0 + m;
    yt[m][col0] = c0[j] + bb0 + y1[(size_t)t * E_ + col0];
    yt[m][col1] = c1[j] + bb1 + y1[(size_t)t * E_ + col1];
  }
  __syncthreads();
#pragma unroll
  for (int rr = 0; rr < 2; ++rr) {
    const int r = 2 * w + rr;
    const int t = t0 + r;
    const int cb = l * 8;
    float vals[8], s = 0.0f, ss = 0.0f;
#pragma unroll
    for (int i = 0; i < 8; ++i) {
      vals[i] = yt[r][cb + i];
      s += vals[i];
      ss += vals[i] * vals[i];
    }
#pragma unroll
    for (int d = 1; d < 32; d <<= 1) { s += __shfl_xor(s, d, 32); ss += __shfl_xor(ss, d, 32); }
    const float mean = s * (1.0f / E_);
    const float var  = ss * (1.0f / E_) - mean * mean;
    const float rstd = rsqrtf(var + LN_EPS_);
#pragma unroll
    for (int i = 0; i < 8; ++i) {
      const int c = cb + i;
      out[(size_t)t * E_ + c] = (vals[i] - mean) * rstd * g2[c] + be2[c];
    }
  }
}

extern "C" void kernel_launch(void* const* d_in, const int* in_sizes, int n_in,
                              void* d_out, int out_size, void* d_ws, size_t ws_size,
                              hipStream_t stream) {
  const float* x   = (const float*)d_in[0];
  const float* Wq  = (const float*)d_in[1];
  const float* Wk  = (const float*)d_in[2];
  const float* Wv  = (const float*)d_in[3];
  const float* Wp  = (const float*)d_in[4];
  const float* bp  = (const float*)d_in[5];
  const float* W1  = (const float*)d_in[6];
  const float* b1  = (const float*)d_in[7];
  const float* W2  = (const float*)d_in[8];
  const float* b2  = (const float*)d_in[9];
  const float* g1  = (const float*)d_in[10];
  const float* be1 = (const float*)d_in[11];
  const float* g2  = (const float*)d_in[12];
  const float* be2 = (const float*)d_in[13];
  float* out = (float*)d_out;

  char* p = (char*)d_ws;
  auto alloc = [&](size_t bytes) -> char* {
    char* r = p;
    p += (bytes + 255) & ~(size_t)255;
    return r;
  };
  bf16* WqT  = (bf16*)alloc((size_t)DK_ * E_ * 2);
  bf16* WkT  = (bf16*)alloc((size_t)DK_ * E_ * 2);
  bf16* WvT  = (bf16*)alloc((size_t)DK_ * E_ * 2);
  bf16* WpfT = (bf16*)alloc((size_t)E_ * DK_ * 2);
  bf16* W1T  = (bf16*)alloc((size_t)FF_ * E_ * 2);
  bf16* W2T  = (bf16*)alloc((size_t)E_ * FF_ * 2);
  bf16* qb   = (bf16*)alloc((size_t)BT_ * DK_ * 2);
  bf16* kb   = (bf16*)alloc((size_t)BT_ * DK_ * 2);
  bf16* vT   = (bf16*)alloc((size_t)BT_ * DK_ * 2);
  bf16* attn = (bf16*)alloc((size_t)BT_ * DK_ * 2);
  float* y1  = (float*)alloc((size_t)BT_ * E_ * 4);
  bf16* h1   = (bf16*)alloc((size_t)BT_ * FF_ * 2);

  // weight prep
  k_transpose<<<dim3((E_ * DK_ + 255) / 256), dim3(256), 0, stream>>>(Wq, WqT, E_, DK_);
  k_transpose<<<dim3((E_ * DK_ + 255) / 256), dim3(256), 0, stream>>>(Wk, WkT, E_, DK_);
  k_transpose<<<dim3((E_ * DK_ + 255) / 256), dim3(256), 0, stream>>>(Wv, WvT, E_, DK_);
  k_transpose<<<dim3((E_ * FF_ + 255) / 256), dim3(256), 0, stream>>>(W1, W1T, E_, FF_);
  k_transpose<<<dim3((FF_ * E_ + 255) / 256), dim3(256), 0, stream>>>(W2, W2T, FF_, E_);
  k_fold_wp<<<dim3((E_ * DK_ + 255) / 256), dim3(256), 0, stream>>>(Wp, WpfT);

  // QKV: one wave per 16-row tile -> BT/16 = 2048 waves, 4 waves/block
  k_qkv<<<dim3(BT_ / 16 / 4), dim3(128), 0, stream>>>(x, WqT, WkT, WvT, qb, kb, vT);

  // causal flash attention
  k_attn<<<dim3(BT_ / 16 / FA_WAVES), dim3(32 * FA_WAVES), 0, stream>>>(qb, kb, vT, attn);

  // proj + residual + LN1
  k_mh_ln<<<dim3(BT_ / 16), dim3(256), 0, stream>>>(attn, WpfT, bp, x, g1, be1, y1);

  // FFN
  k_ff1<<<dim3((BT_ / 16) * 4), dim3(256), 0, stream>>>(y1, W1T, b1, h1);
  k_ff2_ln<<<dim3(BT_ / 16), dim3(256), 0, stream>>>(h1, W2T, b2, y1, g2, be2, out);
}
